// BankedFeedforward_45603962749766
// MI455X (gfx1250) — compile-verified, matrix operand order checked by reading, not run on hardware
//
#include <hip/hip_runtime.h>
#include <hip/hip_bf16.h>

#define D_MODEL   1024
#define D_HIDDEN  4096
#define NB        16
#define NTOK      4096      // 4 * 1024
#define TILE_M    32        // tokens per ffn block tile (full N=1024 per block)
#define CH        64        // hidden chunk
#define KSUP      128       // k super-chunk for W1 LDS staging

typedef __attribute__((ext_vector_type(16))) __bf16 v16bf;
typedef __attribute__((ext_vector_type(8)))  float  v8f;
typedef __attribute__((__vector_size__(16))) int    vi4;   // native int4 vector

#if __has_builtin(__builtin_amdgcn_global_load_async_to_lds_b128) && \
    __has_builtin(__builtin_amdgcn_s_wait_asynccnt)
#define USE_ASYNC_LDS 1
#else
#define USE_ASYNC_LDS 0
#endif

union FragBF {
    v16bf v;
    uint4 q[2];   // two 16-byte halves (ds_load_b128 / global_load_b128 fills)
};

__device__ __forceinline__ unsigned short f2bf(float f) {
    unsigned u = __float_as_uint(f);
    unsigned r = u + 0x7FFFu + ((u >> 16) & 1u);   // round-to-nearest-even
    return (unsigned short)(r >> 16);
}

__device__ __forceinline__ v8f vzero8() {
    v8f v = {0.f, 0.f, 0.f, 0.f, 0.f, 0.f, 0.f, 0.f};
    return v;
}

// ---------------------------------------------------------------------------
// Kernel 0: zero output + expert counters
// ---------------------------------------------------------------------------
__global__ void moe_zero_kernel(float* __restrict__ out, int* __restrict__ counts,
                                size_t n) {
    size_t i = (size_t)blockIdx.x * blockDim.x + threadIdx.x;
    size_t stride = (size_t)gridDim.x * blockDim.x;
    for (; i < n; i += stride) out[i] = 0.f;
    if (blockIdx.x == 0 && threadIdx.x < NB) counts[threadIdx.x] = 0;
}

// ---------------------------------------------------------------------------
// Kernel 1: elementwise fp32 -> bf16 (for X)
// ---------------------------------------------------------------------------
__global__ void moe_cvt_kernel(const float* __restrict__ in,
                               unsigned short* __restrict__ out, size_t n4) {
    size_t i = (size_t)blockIdx.x * blockDim.x + threadIdx.x;
    size_t stride = (size_t)gridDim.x * blockDim.x;
    for (; i < n4; i += stride) {
        float4 v = *(const float4*)(in + 4 * i);
        uint2 pk;
        pk.x = (unsigned)f2bf(v.x) | ((unsigned)f2bf(v.y) << 16);
        pk.y = (unsigned)f2bf(v.z) | ((unsigned)f2bf(v.w) << 16);
        *(uint2*)(out + 4 * i) = pk;
    }
}

// ---------------------------------------------------------------------------
// Kernel 2: batched transpose + fp32->bf16:  in[b][R][C] -> out[b][C][R]
// 32x32 LDS tile keeps both global sides coalesced
// ---------------------------------------------------------------------------
__global__ __launch_bounds__(256)
void moe_tr_kernel(const float* __restrict__ in, unsigned short* __restrict__ out,
                   int R, int C) {
    __shared__ unsigned short tile[32][33];
    int c0 = blockIdx.x * 32, r0 = blockIdx.y * 32, b = blockIdx.z;
    const float* src = in + (size_t)b * R * C;
    int tx = threadIdx.x, ty = threadIdx.y;   // 32 x 8
#pragma unroll
    for (int i = ty; i < 32; i += 8)
        tile[i][tx] = f2bf(src[(size_t)(r0 + i) * C + c0 + tx]);
    __syncthreads();
    unsigned short* dst = out + (size_t)b * C * R;
#pragma unroll
    for (int i = ty; i < 32; i += 8)
        dst[(size_t)(c0 + i) * R + r0 + tx] = tile[tx][i];
}

// ---------------------------------------------------------------------------
// Kernel 3: router — logits, softmax, top-2, build per-expert token lists
// ---------------------------------------------------------------------------
__global__ __launch_bounds__(256)
void moe_router_kernel(const float* __restrict__ X, const float* __restrict__ Wr,
                       const float* __restrict__ br, int* __restrict__ counts,
                       int* __restrict__ lists, float* __restrict__ gates) {
    int t = blockIdx.x * blockDim.x + threadIdx.x;
    if (t >= NTOK) return;

    float logit[NB];
#pragma unroll
    for (int e = 0; e < NB; ++e) logit[e] = br[e];

    const float* xp = X + (size_t)t * D_MODEL;
    for (int d = 0; d < D_MODEL; d += 4) {
        float4 x = *(const float4*)(xp + d);
#pragma unroll
        for (int e = 0; e < NB; ++e) {
            logit[e] += x.x * Wr[(d + 0) * NB + e] + x.y * Wr[(d + 1) * NB + e] +
                        x.z * Wr[(d + 2) * NB + e] + x.w * Wr[(d + 3) * NB + e];
        }
    }
    float mx = logit[0];
#pragma unroll
    for (int e = 1; e < NB; ++e) mx = fmaxf(mx, logit[e]);
    float p[NB], sum = 0.f;
#pragma unroll
    for (int e = 0; e < NB; ++e) { p[e] = __expf(logit[e] - mx); sum += p[e]; }
    float inv = 1.f / sum;
    int i1 = 0; float v1 = p[0];
#pragma unroll
    for (int e = 1; e < NB; ++e) if (p[e] > v1) { v1 = p[e]; i1 = e; }
    int i2 = (i1 == 0) ? 1 : 0; float v2 = p[i2];
#pragma unroll
    for (int e = 0; e < NB; ++e)
        if (e != i1 && p[e] > v2) { v2 = p[e]; i2 = e; }

    int s1 = atomicAdd(&counts[i1], 1);
    lists[i1 * NTOK + s1] = t;
    gates[i1 * NTOK + s1] = v1 * inv;
    int s2 = atomicAdd(&counts[i2], 1);
    lists[i2 * NTOK + s2] = t;
    gates[i2 * NTOK + s2] = v2 * inv;
}

// ---------------------------------------------------------------------------
// Kernel 4: fused expert FFN, bf16 WMMA, full-N tile (no recompute)
//   grid = (m-tiles, expert); 256 threads = 8 wave32
//   wave w: row group wr2 = w&1 (16 of 32 rows), col quarter wq = w>>1
//   first GEMM : Hc[32][64]  — wave computes rows 16*wr2, cols 16*wq
//   second GEMM: Y [32][1024]— wave computes rows 16*wr2, cols 256*wq (16 frags)
//   Weights are pre-converted bf16, pre-transposed: B-fragment = 2 x b128.
// ---------------------------------------------------------------------------
__global__ __launch_bounds__(256)
void moe_ffn_kernel(const unsigned short* __restrict__ Xb,   // [4096][1024] bf16
                    const unsigned short* __restrict__ W1T,  // [e][4096][1024] bf16
                    const unsigned short* __restrict__ W2T,  // [e][1024][4096] bf16
                    const float* __restrict__ b1, const float* __restrict__ b2,
                    const int* __restrict__ counts, const int* __restrict__ lists,
                    const float* __restrict__ gates, float* __restrict__ out) {
    const int e     = blockIdx.y;
    const int mtile = blockIdx.x;
    const int cnt   = counts[e];
    const int m0    = mtile * TILE_M;
    if (m0 >= cnt) return;                 // uniform exit: EXEC all-ones below

    const int tid  = threadIdx.x;
    const int wave = tid >> 5;
    const int lane = tid & 31;
    const int lrow = lane & 15;
    const int half = lane >> 4;
    const int wr2  = wave & 1;             // row group base = 16*wr2
    const int wq   = wave >> 1;            // 0..3

    __shared__ int            Toks[TILE_M];
    __shared__ float          Gts[TILE_M];
    __shared__ unsigned short W1s[CH][KSUP + 8];        // 17408 B
    __shared__ unsigned short Hcs[TILE_M][CH + 8];      // 4608 B

    if (tid < TILE_M) {
        int gidx = m0 + tid;
        bool valid = gidx < cnt;
        Toks[tid] = valid ? lists[e * NTOK + gidx] : 0;
        Gts[tid]  = valid ? gates[e * NTOK + gidx] : 0.f;
    }
    __syncthreads();

    const unsigned short* W1e = W1T + (size_t)e * D_HIDDEN * D_MODEL;
    const unsigned short* W2e = W2T + (size_t)e * D_MODEL * D_HIDDEN;
    const unsigned short* xrow = Xb + (size_t)Toks[16 * wr2 + lrow] * D_MODEL;

    v8f yacc[16];
#pragma unroll
    for (int s = 0; s < 16; ++s) yacc[s] = vzero8();

    for (int hc = 0; hc < D_HIDDEN / CH; ++hc) {
        const int hbase = hc * CH;
        v8f hacc = vzero8();

        for (int ks = 0; ks < D_MODEL / KSUP; ++ks) {
            const int kbase = ks * KSUP;
            __syncthreads();   // previous readers of W1s done
            // --- stage W1T slab [64 rows(h)][128 cols(d)] into LDS ----------
#pragma unroll
            for (int i = 0; i < 4; ++i) {
                int idx = tid + 256 * i;           // 1024 x 16B segments
                int row = idx >> 4;
                int seg = idx & 15;
                const unsigned short* g =
                    W1e + (size_t)(hbase + row) * D_MODEL + kbase + seg * 8;
                unsigned short* l = &W1s[row][seg * 8];
#if USE_ASYNC_LDS
                __builtin_amdgcn_global_load_async_to_lds_b128(
                    (__attribute__((address_space(1))) vi4*)g,
                    (__attribute__((address_space(3))) vi4*)l, 0, 0);
#else
                *(uint4*)l = *(const uint4*)g;
#endif
            }
#if USE_ASYNC_LDS
            __builtin_amdgcn_s_wait_asynccnt(0);
#endif
            __syncthreads();
            // --- first GEMM: Hc += Xchunk @ W1chunk --------------------------
#pragma unroll
            for (int kk = 0; kk < KSUP / 32; ++kk) {
                FragBF a;   // A from global bf16 X (gathered row, L2 resident)
                const unsigned short* pa = xrow + kbase + kk * 32 + 8 * half;
                a.q[0] = *(const uint4*)pa;
                a.q[1] = *(const uint4*)(pa + 16);
                FragBF b;   // B from LDS (shared by both row-group waves)
                const unsigned short* pb = &W1s[16 * wq + lrow][kk * 32 + 8 * half];
                b.q[0] = *(const uint4*)pb;
                b.q[1] = *(const uint4*)(pb + 16);
                hacc = __builtin_amdgcn_wmma_f32_16x16x32_bf16(
                    false, a.v, false, b.v, (short)0, hacc, false, false);
            }
        }
        // --- bias + relu -> Hcs (C/D layout: row = r + 8*half) --------------
        {
            int ncol = 16 * wq + lrow;                       // 0..63
            float bb = b1[(size_t)e * D_HIDDEN + hbase + ncol];
#pragma unroll
            for (int r = 0; r < 8; ++r) {
                float v = hacc[r] + bb;
                v = v > 0.f ? v : 0.f;
                Hcs[16 * wr2 + 8 * half + r][ncol] = f2bf(v);
            }
        }
        __syncthreads();
        // --- second GEMM: Y += Hc @ W2chunk, B direct from global bf16 ------
#pragma unroll
        for (int kk = 0; kk < CH / 32; ++kk) {
            FragBF a;
            const unsigned short* pa = &Hcs[16 * wr2 + lrow][kk * 32 + 8 * half];
            a.q[0] = *(const uint4*)pa;
            a.q[1] = *(const uint4*)(pa + 16);
#pragma unroll
            for (int s = 0; s < 16; ++s) {
                int n = 256 * wq + 16 * s + lrow;
                FragBF b;
                const unsigned short* pb =
                    W2e + (size_t)n * D_HIDDEN + hbase + kk * 32 + 8 * half;
                b.q[0] = *(const uint4*)pb;
                b.q[1] = *(const uint4*)(pb + 16);
                yacc[s] = __builtin_amdgcn_wmma_f32_16x16x32_bf16(
                    false, a.v, false, b.v, (short)0, yacc[s], false, false);
            }
        }
        __syncthreads();   // Hcs reads done before next chunk rewrites it
    }

    // --- epilogue: out[token] += gate * (y + b2) ----------------------------
    int rbase = 16 * wr2 + 8 * half;
    float gate[8];
    int   tok[8];
#pragma unroll
    for (int r = 0; r < 8; ++r) {
        gate[r] = Gts[rbase + r];          // 0 for padded rows -> adds 0
        tok[r]  = Toks[rbase + r];
    }
#pragma unroll
    for (int s = 0; s < 16; ++s) {
        int colg   = 256 * wq + 16 * s + lrow;
        float bias = b2[(size_t)e * D_MODEL + colg];
#pragma unroll
        for (int r = 0; r < 8; ++r) {
            float v = gate[r] * (yacc[s][r] + bias);
            atomicAdd(&out[(size_t)tok[r] * D_MODEL + colg], v);
        }
    }
}

// ---------------------------------------------------------------------------
extern "C" void kernel_launch(void* const* d_in, const int* in_sizes, int n_in,
                              void* d_out, int out_size, void* d_ws, size_t ws_size,
                              hipStream_t stream) {
    const float* X  = (const float*)d_in[0];
    const float* Wr = (const float*)d_in[1];
    const float* br = (const float*)d_in[2];
    const float* W1 = (const float*)d_in[3];
    const float* b1 = (const float*)d_in[4];
    const float* W2 = (const float*)d_in[5];
    const float* b2 = (const float*)d_in[6];
    float* out = (float*)d_out;

    // workspace layout (assumes ws_size >= ~265 MB; small vs 536 MB of inputs)
    char*  ws     = (char*)d_ws;
    size_t off    = 0;
    int*   counts = (int*)(ws + off);             off += 256;
    int*   lists  = (int*)(ws + off);             off += (size_t)NB * NTOK * 4;
    float* gates  = (float*)(ws + off);           off += (size_t)NB * NTOK * 4;
    off = (off + 255) & ~(size_t)255;
    unsigned short* Xb  = (unsigned short*)(ws + off);
    off += (size_t)NTOK * D_MODEL * 2;
    unsigned short* W1T = (unsigned short*)(ws + off);   // [e][4096][1024]
    off += (size_t)NB * D_HIDDEN * D_MODEL * 2;
    unsigned short* W2T = (unsigned short*)(ws + off);   // [e][1024][4096]

    moe_zero_kernel<<<256, 256, 0, stream>>>(out, counts, (size_t)NTOK * D_MODEL);
    moe_cvt_kernel<<<512, 256, 0, stream>>>(X, Xb, (size_t)NTOK * D_MODEL / 4);
    {   // W1[e][1024][4096] -> W1T[e][4096][1024]
        dim3 g(D_HIDDEN / 32, D_MODEL / 32, NB);
        moe_tr_kernel<<<g, dim3(32, 8), 0, stream>>>(W1, W1T, D_MODEL, D_HIDDEN);
    }
    {   // W2[e][4096][1024] -> W2T[e][1024][4096]
        dim3 g(D_MODEL / 32, D_HIDDEN / 32, NB);
        moe_tr_kernel<<<g, dim3(32, 8), 0, stream>>>(W2, W2T, D_HIDDEN, D_MODEL);
    }
    moe_router_kernel<<<NTOK / 256, 256, 0, stream>>>(X, Wr, br, counts, lists, gates);

    dim3 grid(NTOK / TILE_M, NB);   // mtile fastest => per-expert L2 reuse
    moe_ffn_kernel<<<grid, 256, 0, stream>>>(Xb, W1T, W2T, b1, b2,
                                             counts, lists, gates, out);
}
